// HGT_44933947851300
// MI455X (gfx1250) — matrix operand order repaired and checked by reading, compiled, not run on hardware
//
#include <hip/hip_runtime.h>
#include <math.h>

typedef __attribute__((ext_vector_type(16))) _Float16 v16h;
typedef __attribute__((ext_vector_type(8)))  float    v8f;

#define FLAG_RELU 1

__device__ __forceinline__ float fast_exp(float x) { return __expf(x); }

__device__ __forceinline__ float gelu_fast(float x) {
  // tanh-approx gelu; tanh(u) = 1 - 2/(1+exp(2u)) (branchless: v_exp + v_rcp)
  float u = 0.7978845608028654f * (x + 0.044715f * x * x * x);
  float t = 1.0f - 2.0f * __builtin_amdgcn_rcpf(1.0f + fast_exp(2.0f * u));
  return 0.5f * x * (1.0f + t);
}

// ---------------------------------------------------------------------------
// Weight pre-pack: f32 [K][N] -> f16 WMMA B fragments in this file's contiguous k order (same as A).
// Fragment (kc, nc) holds, for lane: column nc*16+(lane&15),
// K rows kc*32 + (lane>>4)*16 + j, j=0..15 -> 16 contiguous halves (32B).
// Offset of a fragment lane-chunk: ((kc*ntiles + nc)*32 + lane)*16 halves.
// ---------------------------------------------------------------------------
__global__ void pack_b_kernel(const float* __restrict__ B, _Float16* __restrict__ out,
                              int Kdim, int N, long inZ, long outZ)
{
  int idx = blockIdx.x * blockDim.x + threadIdx.x;
  int ntiles = N >> 4;
  int kchunks = Kdim >> 5;
  int total = kchunks * ntiles * 32;
  if (idx >= total) return;
  int lane = idx & 31;
  int nc = (idx >> 5) % ntiles;
  int kc = (idx >> 5) / ntiles;
  const float* Bz = B + (size_t)blockIdx.z * inZ;
  _Float16* oz = out + (size_t)blockIdx.z * outZ + (size_t)idx * 16;
  int col = nc * 16 + (lane & 15);
  int krow = kc * 32 + (lane >> 4) * 16;
  v16h frag;
  #pragma unroll
  for (int j = 0; j < 16; ++j)
    frag[j] = (_Float16)Bz[(size_t)(krow + j) * N + col];
  *(v16h*)oz = frag;
}

// ---------------------------------------------------------------------------
// Generic WMMA GEMM, one wave per block, 16(M) x 16*NSUB(N) output tile:
//   C[z] = epilogue( A[z][M,K] @ B[z][K,N] + bias )
// A: f32 row-major (converted to f16 in-register; GEL: compile-time gelu).
// B: pre-packed f16 fragments (see pack_b_kernel); one 32B load per fragment.
// z = blockIdx.z selects a head slice via aZ/bZ/cZ strides.
// ---------------------------------------------------------------------------
template <int NSUB, bool GEL>
__global__ __launch_bounds__(32)
void wmma_gemm(const float* __restrict__ A, int lda, long aZ,
               const _Float16* __restrict__ B, int ntiles, long bZ,
               float* __restrict__ C, int ldc, long cZ,
               const float* __restrict__ bias,
               const float* __restrict__ resid,
               const float* __restrict__ skipPtr, int skipIdx,
               int K, int flags)
{
  const int lane = threadIdx.x & 31;
  const int half = lane >> 4;
  const int l16  = lane & 15;
  const long z   = blockIdx.z;
  const float*    Ap = A + z * aZ;
  const _Float16* Bp = B + z * bZ;
  float* Cp = C + z * cZ;
  const int m0 = blockIdx.x << 4;
  const int n0 = blockIdx.y * (16 * NSUB);

  const int arow = m0 + l16;
  const int ako  = half * 16;

  v8f zero = {};
  v8f acc[NSUB];
  #pragma unroll
  for (int i = 0; i < NSUB; ++i) acc[i] = zero;

  for (int k = 0; k < K; k += 32) {
    // A fragment: one contiguous 16-float run per lane -> 4x b128 loads.
    const float* ap = Ap + (size_t)arow * lda + k;
    float4 a0 = *(const float4*)(ap + ako);
    float4 a1 = *(const float4*)(ap + ako + 4);
    float4 a2 = *(const float4*)(ap + ako + 8);
    float4 a3 = *(const float4*)(ap + ako + 12);
    float av[16] = { a0.x, a0.y, a0.z, a0.w, a1.x, a1.y, a1.z, a1.w,
                     a2.x, a2.y, a2.z, a2.w, a3.x, a3.y, a3.z, a3.w };
    v16h af;
    #pragma unroll
    for (int j = 0; j < 16; ++j) {
      float v = av[j];
      if (GEL) v = gelu_fast(v);
      af[j] = (_Float16)v;
    }
    // NSUB pre-packed B fragments; each is one contiguous 32B load.
    const size_t kc = (size_t)(k >> 5);
    #pragma unroll
    for (int i = 0; i < NSUB; ++i) {
      const size_t nIdx = (size_t)blockIdx.y * NSUB + i;
      v16h bf = *(const v16h*)(Bp + (((kc * ntiles + nIdx) * 32 + lane) << 4));
      acc[i] = __builtin_amdgcn_wmma_f32_16x16x32_f16(
          false, af, false, bf, (short)0, acc[i], false, false);
    }
  }

  // Epilogue
  float beta = 0.0f, ombeta = 0.0f;
  if (resid) {
    float sv = skipPtr[skipIdx];
    beta = __builtin_amdgcn_rcpf(1.0f + fast_exp(-sv));  // sigmoid
    ombeta = 1.0f - beta;
  }
  const int mrow0 = m0 + half * 8;
  #pragma unroll
  for (int i = 0; i < NSUB; ++i) {
    const int bn = n0 + i * 16 + l16;
    float bv = bias ? bias[bn] : 0.0f;
    #pragma unroll
    for (int r = 0; r < 8; ++r) {
      float v = acc[i][r] + bv;
      if (flags & FLAG_RELU) v = fmaxf(v, 0.0f);
      size_t off = (size_t)(mrow0 + r) * ldc + bn;
      if (resid) v = beta * v + ombeta * resid[off];
      Cp[off] = v;
    }
  }
}

__global__ void fill_kernel(float* __restrict__ p, float v, long n) {
  long i = (long)blockIdx.x * blockDim.x + threadIdx.x;
  if (i < n) p[i] = v;
}

// Monotonic-encoding float atomic max (buffer pre-filled with -inf).
__device__ __forceinline__ void atomicMaxFloat(float* addr, float val) {
  if (val >= 0.0f) atomicMax((int*)addr, __float_as_int(val));
  else             atomicMin((unsigned int*)addr, __float_as_uint(val));
}

// alpha[e,h] = dot(kt[src,h,:], q[dst,h,:]) * p_rel[h] / sqrt(D); amax via atomics
__global__ void edge_alpha_kernel(const int* __restrict__ src, const int* __restrict__ dst,
                                  const float* __restrict__ kt,     // [Ns, H*D]
                                  const float* __restrict__ kqv_d,  // [Nd, 3C], Q at +C
                                  const float* __restrict__ prel,   // [H]
                                  float* __restrict__ alpha,        // [E, H]
                                  float* __restrict__ amax,         // [Nd, H]
                                  int E)
{
  int t = blockIdx.x * blockDim.x + threadIdx.x;
  if (t >= E * 4) return;
  int e = t >> 2, h = t & 3;
  int s = src[e], d = dst[e];
  const float4* kp = (const float4*)(kt    + (size_t)s * 128 + h * 32);
  const float4* qp = (const float4*)(kqv_d + (size_t)d * 384 + 128 + h * 32);
  float acc = 0.0f;
  #pragma unroll
  for (int j = 0; j < 8; ++j) {
    float4 a = kp[j], b = qp[j];
    acc += a.x * b.x + a.y * b.y + a.z * b.z + a.w * b.w;
  }
  float al = acc * prel[h] * 0.17677669529663687f; // 1/sqrt(32)
  alpha[t] = al;
  atomicMaxFloat(amax + (size_t)d * 4 + h, al);
}

// alpha[e,h] = exp(alpha - amax[dst]); denom[dst,h] += alpha
__global__ void edge_exp_kernel(const int* __restrict__ dst,
                                float* __restrict__ alpha,
                                const float* __restrict__ amax,
                                float* __restrict__ denom, int E)
{
  int t = blockIdx.x * blockDim.x + threadIdx.x;
  if (t >= E * 4) return;
  int e = t >> 2, h = t & 3;
  int d = dst[e];
  float ex = fast_exp(alpha[t] - amax[(size_t)d * 4 + h]);
  alpha[t] = ex;
  atomicAdd(denom + (size_t)d * 4 + h, ex);
}

// agg[dst,h,:] += vt[src,h,:] * alpha/(denom+eps)
__global__ void edge_agg_kernel(const int* __restrict__ src, const int* __restrict__ dst,
                                const float* __restrict__ vt,
                                const float* __restrict__ alpha,
                                const float* __restrict__ denom,
                                float* __restrict__ agg, int E)
{
  int t = blockIdx.x * blockDim.x + threadIdx.x;
  if (t >= E * 4) return;
  int e = t >> 2, h = t & 3;
  int s = src[e], d = dst[e];
  float w = alpha[t] / (denom[(size_t)d * 4 + h] + 1e-16f);
  const float4* vp = (const float4*)(vt + (size_t)s * 128 + h * 32);
  float*        ap = agg + (size_t)d * 128 + h * 32;
  #pragma unroll
  for (int j = 0; j < 8; ++j) {
    float4 v = vp[j];
    atomicAdd(ap + 4 * j + 0, v.x * w);
    atomicAdd(ap + 4 * j + 1, v.y * w);
    atomicAdd(ap + 4 * j + 2, v.z * w);
    atomicAdd(ap + 4 * j + 3, v.w * w);
  }
}

extern "C" void kernel_launch(void* const* d_in, const int* in_sizes, int n_in,
                              void* d_out, int out_size, void* d_ws, size_t ws_size,
                              hipStream_t stream)
{
  (void)n_in; (void)out_size; (void)ws_size;
  const float* x_paper  = (const float*)d_in[0];
  const float* x_author = (const float*)d_in[1];
  const float* Wp_in = (const float*)d_in[2];
  const float* bp_in = (const float*)d_in[3];
  const float* Wa_in = (const float*)d_in[4];
  const float* ba_in = (const float*)d_in[5];
  const float* kqv_W = (const float*)d_in[6];
  const float* kqv_b = (const float*)d_in[7];
  const float* out_W = (const float*)d_in[8];
  const float* out_b = (const float*)d_in[9];
  const float* skip  = (const float*)d_in[10];
  const float* a_rel = (const float*)d_in[11];
  const float* m_rel = (const float*)d_in[12];
  const float* p_rel = (const float*)d_in[13];
  const int* e_src[3] = { (const int*)d_in[14], (const int*)d_in[16], (const int*)d_in[18] };
  const int* e_dst[3] = { (const int*)d_in[15], (const int*)d_in[17], (const int*)d_in[19] };

  const int NP = in_sizes[0] / 256;   // 50000 (multiple of 16)
  const int NA = in_sizes[1] / 128;   // 20000 (multiple of 16)
  const int E[3] = { in_sizes[14], in_sizes[16], in_sizes[18] };
  const int Cc = 128, H = 4, L = 2;

  // ---- workspace carve (floats, 1KB-aligned chunks) ----
  float* ws = (float*)d_ws;
  size_t cur = 0;
  auto alloc = [&](size_t n) {
    float* p = ws + cur;
    cur += (n + 255) & ~(size_t)255;
    return p;
  };
  float* xs0  = alloc((size_t)NP * Cc);
  float* xs1  = alloc((size_t)NA * Cc);
  float* kqv0 = alloc((size_t)NP * 3 * Cc);
  float* kqv1 = alloc((size_t)NA * 3 * Cc);
  float* tbuf = alloc((size_t)NP * Cc);          // kt/vt scratch (NP >= NA)
  float* alphab[3];
  for (int r = 0; r < 3; ++r) alphab[r] = alloc((size_t)E[r] * H);
  float* amaxT[2] = { alloc((size_t)NP * H), alloc((size_t)NA * H) };
  float* denT [2] = { alloc((size_t)NP * H), alloc((size_t)NA * H) };
  float* aggT [2] = { alloc((size_t)NP * Cc), alloc((size_t)NA * Cc) };
  // packed f16 weights (sizes in halves -> alloc halves/2 floats)
  _Float16* pWp  = (_Float16*)alloc((256 / 32) * (128 / 16) * 512 / 2);        // 256x128
  _Float16* pWa  = (_Float16*)alloc((128 / 32) * (128 / 16) * 512 / 2);        // 128x128
  const long zKqv = (long)(128 / 32) * (384 / 16) * 512;                       // per (l,t)
  _Float16* pKqv = (_Float16*)alloc((size_t)4 * zKqv / 2);
  const long zOut = (long)(128 / 32) * (128 / 16) * 512;                       // per (l,t)
  _Float16* pOut = (_Float16*)alloc((size_t)4 * zOut / 2);
  const long zRel = (long)(32 / 32) * (32 / 16) * 512;                         // per head: 1024
  _Float16* pA   = (_Float16*)alloc((size_t)(L * 3 * H) * zRel / 2);
  _Float16* pM   = (_Float16*)alloc((size_t)(L * 3 * H) * zRel / 2);

  const int Nt[2] = { NP, NA };
  float* xsT[2]   = { xs0, xs1 };
  float* kqvT[2]  = { kqv0, kqv1 };
  const int rel_s[3] = { 0, 1, 0 };  // cites, writes, written_by
  const int rel_d[3] = { 0, 0, 1 };

  const float* fnull = nullptr;
  dim3 blk(32, 1, 1);

  // ---- weight pre-pack (f32 -> fragment-swizzled f16) ----
  pack_b_kernel<<<dim3((2048 + 255) / 256, 1, 1), dim3(256), 0, stream>>>(
      Wp_in, pWp, 256, 128, 0L, 0L);
  pack_b_kernel<<<dim3((1024 + 255) / 256, 1, 1), dim3(256), 0, stream>>>(
      Wa_in, pWa, 128, 128, 0L, 0L);
  pack_b_kernel<<<dim3(((int)zKqv / 16 + 255) / 256, 1, 4), dim3(256), 0, stream>>>(
      kqv_W, pKqv, 128, 384, (long)Cc * 3 * Cc, zKqv);
  pack_b_kernel<<<dim3(((int)zOut / 16 + 255) / 256, 1, 4), dim3(256), 0, stream>>>(
      out_W, pOut, 128, 128, (long)Cc * Cc, zOut);
  pack_b_kernel<<<dim3(1, 1, L * 3 * H), dim3(256), 0, stream>>>(
      a_rel, pA, 32, 32, 1024L, zRel);
  pack_b_kernel<<<dim3(1, 1, L * 3 * H), dim3(256), 0, stream>>>(
      m_rel, pM, 32, 32, 1024L, zRel);

  // ---- input projections + relu  (N=128 -> grid.y = 2, ntiles = 8) ----
  wmma_gemm<4, false><<<dim3(NP / 16, 2, 1), blk, 0, stream>>>(
      x_paper, 256, 0L, pWp, 8, 0L, xs0, Cc, 0L,
      bp_in, fnull, fnull, 0, 256, FLAG_RELU);
  wmma_gemm<4, false><<<dim3(NA / 16, 2, 1), blk, 0, stream>>>(
      x_author, 128, 0L, pWa, 8, 0L, xs1, Cc, 0L,
      ba_in, fnull, fnull, 0, 128, FLAG_RELU);

  for (int l = 0; l < L; ++l) {
    // kqv projections: [N,128] @ [128,384] + b  (ntiles = 24, grid.y = 6)
    for (int t = 0; t < 2; ++t) {
      wmma_gemm<4, false><<<dim3(Nt[t] / 16, 6, 1), blk, 0, stream>>>(
          xsT[t], Cc, 0L,
          pKqv + (size_t)(l * 2 + t) * zKqv, 24, 0L,
          kqvT[t], 3 * Cc, 0L,
          kqv_b + (size_t)(l * 2 + t) * 3 * Cc,
          fnull, fnull, 0, Cc, 0);
    }
    // init softmax / aggregation buffers
    for (int t = 0; t < 2; ++t) {
      long nm = (long)Nt[t] * H, nc = (long)Nt[t] * Cc;
      fill_kernel<<<dim3((nm + 255) / 256), dim3(256), 0, stream>>>(amaxT[t], -INFINITY, nm);
      fill_kernel<<<dim3((nm + 255) / 256), dim3(256), 0, stream>>>(denT[t], 0.0f, nm);
      fill_kernel<<<dim3((nc + 255) / 256), dim3(256), 0, stream>>>(aggT[t], 0.0f, nc);
    }
    // pass 1: per relation, node-level K transform (per-head WMMA, ntiles=2, NSUB=2)
    //         then edge alpha + atomic amax
    for (int r = 0; r < 3; ++r) {
      int s = rel_s[r], d = rel_d[r];
      wmma_gemm<2, false><<<dim3(Nt[s] / 16, 1, H), blk, 0, stream>>>(
          kqvT[s], 3 * Cc, 32L,                               // K slice, head stride 32
          pA + (size_t)((l * 3 + r) * H) * zRel, 2, zRel,
          tbuf, Cc, 32L,
          fnull, fnull, fnull, 0, 32, 0);
      int nthr = E[r] * H;
      edge_alpha_kernel<<<dim3((nthr + 255) / 256), dim3(256), 0, stream>>>(
          e_src[r], e_dst[r], tbuf, kqvT[d],
          p_rel + (size_t)(l * 3 + r) * H,
          alphab[r], amaxT[d], E[r]);
    }
    // pass 2: exp + denominators (needs all relations' amax complete)
    for (int r = 0; r < 3; ++r) {
      int d = rel_d[r];
      int nthr = E[r] * H;
      edge_exp_kernel<<<dim3((nthr + 255) / 256), dim3(256), 0, stream>>>(
          e_dst[r], alphab[r], amaxT[d], denT[d], E[r]);
    }
    // pass 3: node-level V transform + weighted scatter aggregation
    for (int r = 0; r < 3; ++r) {
      int s = rel_s[r], d = rel_d[r];
      wmma_gemm<2, false><<<dim3(Nt[s] / 16, 1, H), blk, 0, stream>>>(
          kqvT[s] + 2 * Cc, 3 * Cc, 32L,                      // V slice
          pM + (size_t)((l * 3 + r) * H) * zRel, 2, zRel,
          tbuf, Cc, 32L,
          fnull, fnull, fnull, 0, 32, 0);
      int nthr = E[r] * H;
      edge_agg_kernel<<<dim3((nthr + 255) / 256), dim3(256), 0, stream>>>(
          e_src[r], e_dst[r], tbuf, alphab[r], denT[d], aggT[d], E[r]);
    }
    // out projection: gelu(agg) @ out_W + b, sigmoid-gated residual; last layer -> d_out
    for (int t = 0; t < 2; ++t) {
      float* Ctgt = (l == L - 1)
                        ? ((float*)d_out + (t ? (size_t)NP * Cc : 0))
                        : xsT[t];
      wmma_gemm<4, true><<<dim3(Nt[t] / 16, 2, 1), blk, 0, stream>>>(
          aggT[t], Cc, 0L,
          pOut + (size_t)(l * 2 + t) * zOut, 8, 0L,
          Ctgt, Cc, 0L,
          out_b + (size_t)(l * 2 + t) * Cc,
          xsT[t], skip, l * 2 + t,
          Cc, 0);
    }
  }
}
